// MMDLoss_61847529063061
// MI455X (gfx1250) — matrix operand order repaired and checked by reading, not compiled, not run on hardware
//
#include <hip/hip_runtime.h>

typedef __attribute__((ext_vector_type(16))) __bf16 v16bf;
typedef __attribute__((ext_vector_type(8)))  float  v8f;

#define NROW  1024
#define BATCH 512
#define CHN   32
#define DIM   768
#define KS    32
#define LSTR  40   // padded LDS row stride in bf16 elems (80 B -> every row 16B-aligned)

__device__ __forceinline__ unsigned short f2bf(float x) {
  unsigned u = __float_as_uint(x);
  u += 0x7FFFu + ((u >> 16) & 1u);          // round-to-nearest-even to bf16
  return (unsigned short)(u >> 16);
}

__device__ __forceinline__ void split_bf(float x, unsigned short& h, unsigned short& l) {
  unsigned short hb = f2bf(x);
  float hf = __uint_as_float(((unsigned)hb) << 16);
  h = hb;
  l = f2bf(x - hf);                          // residual in bf16 -> ~2^-16 rel err total
}

__device__ __forceinline__ const float* row_ptr(const float* s, const float* t, int i, int c) {
  return (i < BATCH) ? (s + ((size_t)i * CHN + c) * DIM)
                     : (t + ((size_t)(i - BATCH) * CHN + c) * DIM);
}

union Frag { v16bf v; uint4 q[2]; };

// ---------------- pass 0: per-row squared norms ----------------
__global__ __launch_bounds__(256) void mmd_sq(const float* __restrict__ src,
                                              const float* __restrict__ tgt,
                                              float* __restrict__ sq) {
  int wid = threadIdx.x >> 5, lane = threadIdx.x & 31;
  int g = blockIdx.x * 8 + wid;              // 0..32767 over (i,c)
  int i = g >> 5, c = g & 31;
  const float4* p = (const float4*)row_ptr(src, tgt, i, c);
  float s = 0.f;
#pragma unroll
  for (int t = 0; t < 6; ++t) {              // 6 * 32 lanes * 4 = 768
    float4 v = p[lane + t * 32];
    s += v.x * v.x + v.y * v.y + v.z * v.z + v.w * v.w;
  }
#pragma unroll
  for (int off = 16; off > 0; off >>= 1) s += __shfl_xor(s, off, 32);
  if (lane == 0) sq[c * NROW + i] = s;
}

// ---------------- GEMM + epilogue (pass1: sum d2, pass3: exp kernel) ----------------
template <bool PASS3>
__global__ __launch_bounds__(256) void mmd_gemm(const float* __restrict__ src,
                                                const float* __restrict__ tgt,
                                                const float* __restrict__ sq,
                                                const float* __restrict__ invbw,
                                                float* __restrict__ part) {
  __shared__ unsigned short sAhi[128 * LSTR];
  __shared__ unsigned short sAlo[128 * LSTR];
  __shared__ unsigned short sBhi[128 * LSTR];
  __shared__ unsigned short sBlo[128 * LSTR];
  __shared__ float wsum[8];

  const int tid  = threadIdx.x;
  const int lane = tid & 31, wid = tid >> 5;
  const int tn = blockIdx.x, tm = blockIdx.y, c = blockIdx.z;

  // loader mapping: 2 threads per row, 16 k-floats each
  const int lr = tid >> 1;
  const int lk = (tid & 1) << 4;
  const float* aptr = row_ptr(src, tgt, tm * 128 + lr, c) + lk;
  const float* bptr = row_ptr(src, tgt, tn * 128 + lr, c) + lk;

  // wave tile mapping: 2x4 waves, each 64(M) x 32(N)
  const int waveM = wid & 1, waveN = wid >> 1;
  const int half = lane >> 4, l15 = lane & 15;

  v8f zero = {0.f, 0.f, 0.f, 0.f, 0.f, 0.f, 0.f, 0.f};
  v8f acc[4][2];
#pragma unroll
  for (int m = 0; m < 4; ++m) {
#pragma unroll
    for (int n = 0; n < 2; ++n) acc[m][n] = zero;
  }

  for (int ks = 0; ks < DIM; ks += KS) {
    // ---- stage fp32 -> (hi,lo) bf16 slab into LDS ----
    const float4* a4 = (const float4*)(aptr + ks);
    const float4* b4 = (const float4*)(bptr + ks);
#pragma unroll
    for (int q = 0; q < 4; ++q) {
      float4 va = a4[q];
      float4 vb = b4[q];
      int base = lr * LSTR + lk + q * 4;
      unsigned short h0, h1, h2, h3, l0, l1, l2, l3;
      split_bf(va.x, h0, l0); split_bf(va.y, h1, l1);
      split_bf(va.z, h2, l2); split_bf(va.w, h3, l3);
      *(unsigned*)&sAhi[base]     = (unsigned)h0 | ((unsigned)h1 << 16);
      *(unsigned*)&sAhi[base + 2] = (unsigned)h2 | ((unsigned)h3 << 16);
      *(unsigned*)&sAlo[base]     = (unsigned)l0 | ((unsigned)l1 << 16);
      *(unsigned*)&sAlo[base + 2] = (unsigned)l2 | ((unsigned)l3 << 16);
      split_bf(vb.x, h0, l0); split_bf(vb.y, h1, l1);
      split_bf(vb.z, h2, l2); split_bf(vb.w, h3, l3);
      *(unsigned*)&sBhi[base]     = (unsigned)h0 | ((unsigned)h1 << 16);
      *(unsigned*)&sBhi[base + 2] = (unsigned)h2 | ((unsigned)h3 << 16);
      *(unsigned*)&sBlo[base]     = (unsigned)l0 | ((unsigned)l1 << 16);
      *(unsigned*)&sBlo[base + 2] = (unsigned)l2 | ((unsigned)l3 << 16);
    }
    __syncthreads();

    // ---- B fragments: two 16B chunks per lane (column = l15, K = half*8 and 16 + half*8) ----
    Frag bh[2], bl[2];
#pragma unroll
    for (int n16 = 0; n16 < 2; ++n16) {
      int off = (waveN * 32 + n16 * 16 + l15) * LSTR + half * 8;
      bh[n16].q[0] = *(const uint4*)&sBhi[off];
      bh[n16].q[1] = *(const uint4*)&sBhi[off + 16];
      bl[n16].q[0] = *(const uint4*)&sBlo[off];
      bl[n16].q[1] = *(const uint4*)&sBlo[off + 16];
    }
    // ---- A fragments + 3-way split WMMA ----
#pragma unroll
    for (int m16 = 0; m16 < 4; ++m16) {
      int ro = (waveM * 64 + m16 * 16 + l15) * LSTR;
      Frag ah, al;
      ah.q[0] = *(const uint4*)&sAhi[ro + half * 8];
      ah.q[1] = *(const uint4*)&sAhi[ro + 16 + half * 8];
      al.q[0] = *(const uint4*)&sAlo[ro + half * 8];
      al.q[1] = *(const uint4*)&sAlo[ro + 16 + half * 8];
#pragma unroll
      for (int n16 = 0; n16 < 2; ++n16) {
        acc[m16][n16] = __builtin_amdgcn_wmma_f32_16x16x32_bf16(
            false, ah.v, false, bh[n16].v, (short)0, acc[m16][n16], false, false);
        acc[m16][n16] = __builtin_amdgcn_wmma_f32_16x16x32_bf16(
            false, ah.v, false, bl[n16].v, (short)0, acc[m16][n16], false, false);
        acc[m16][n16] = __builtin_amdgcn_wmma_f32_16x16x32_bf16(
            false, al.v, false, bh[n16].v, (short)0, acc[m16][n16], false, false);
      }
    }
    __syncthreads();
  }

  // ---- epilogue: d2 = sq_i + sq_j - 2G, then sum (pass1) or RBF sum (pass3) ----
  float ibw0 = 0.f, ibw1 = 0.f, ibw2 = 0.f, ibw3 = 0.f, ibw4 = 0.f;
  if (PASS3) {
    ibw0 = invbw[c * 5 + 0]; ibw1 = invbw[c * 5 + 1]; ibw2 = invbw[c * 5 + 2];
    ibw3 = invbw[c * 5 + 3]; ibw4 = invbw[c * 5 + 4];
  }
  const float* sqc = sq + c * NROW;
  float local = 0.f;
#pragma unroll
  for (int m16 = 0; m16 < 4; ++m16) {
    int ib = tm * 128 + waveM * 64 + m16 * 16 + half * 8;
    float sqi[8];
#pragma unroll
    for (int r = 0; r < 8; ++r) sqi[r] = sqc[ib + r];
#pragma unroll
    for (int n16 = 0; n16 < 2; ++n16) {
      int j = tn * 128 + waveN * 32 + n16 * 16 + l15;
      float sqj = sqc[j];
#pragma unroll
      for (int r = 0; r < 8; ++r) {
        float d2 = sqi[r] + sqj - 2.0f * acc[m16][n16][r];
        if (PASS3) {
          local += __expf(-d2 * ibw0) + __expf(-d2 * ibw1) + __expf(-d2 * ibw2) +
                   __expf(-d2 * ibw3) + __expf(-d2 * ibw4);
        } else {
          local += d2;
        }
      }
    }
  }
#pragma unroll
  for (int off = 16; off > 0; off >>= 1) local += __shfl_xor(local, off, 32);
  if (lane == 0) wsum[wid] = local;
  __syncthreads();
  if (tid == 0) {
    float s = 0.f;
#pragma unroll
    for (int w = 0; w < 8; ++w) s += wsum[w];
    part[c * 64 + tm * 8 + tn] = s;   // one fixed slot per tile -> deterministic
  }
}

// ---------------- pass 2: bandwidths ----------------
__global__ void mmd_bw(const float* __restrict__ part1, float* __restrict__ invbw) {
  int c = threadIdx.x;
  if (c < CHN) {
    float s = 0.f;
    for (int t = 0; t < 64; ++t) s += part1[c * 64 + t];
    float bw = s / ((float)NROW * (float)NROW - (float)NROW) / 4.0f;  // / 2^(5/2 floor)
    float b = bw;
    for (int k = 0; k < 5; ++k) { invbw[c * 5 + k] = 1.0f / b; b *= 2.0f; }
  }
}

// ---------------- pass 4: signed quadrant combine + channel mean ----------------
__global__ void mmd_final(const float* __restrict__ part3, float* __restrict__ out) {
  __shared__ float ch[CHN];
  int c = threadIdx.x;
  if (c < CHN) {
    float s = 0.f;
    for (int t = 0; t < 64; ++t) {
      int tmq = (t >> 3) >> 2;   // tile row quadrant (0: i<512, 1: i>=512)
      int tnq = (t & 7) >> 2;    // tile col quadrant
      float sg = (tmq == tnq) ? 1.0f : -1.0f;  // XX,YY plus; XY,YX minus
      s += sg * part3[c * 64 + t];
    }
    ch[c] = s * (1.0f / (512.0f * 512.0f));    // each quadrant mean over B*B
  }
  __syncthreads();
  if (c == 0) {
    float t = 0.f;
    for (int k = 0; k < CHN; ++k) t += ch[k];
    out[0] = t / 32.0f;
  }
}

extern "C" void kernel_launch(void* const* d_in, const int* in_sizes, int n_in,
                              void* d_out, int out_size, void* d_ws, size_t ws_size,
                              hipStream_t stream) {
  (void)in_sizes; (void)n_in; (void)out_size; (void)ws_size;
  const float* src = (const float*)d_in[0];
  const float* tgt = (const float*)d_in[1];
  float* out = (float*)d_out;

  float* ws    = (float*)d_ws;
  float* sq    = ws;                          // 32*1024 floats
  float* part1 = ws + 32768;                  // 32*64
  float* invbw = ws + 32768 + 2048;           // 32*5
  float* part3 = ws + 32768 + 2048 + 160;     // 32*64   (~148 KB total)

  mmd_sq<<<dim3(4096), dim3(256), 0, stream>>>(src, tgt, sq);
  mmd_gemm<false><<<dim3(8, 8, 32), dim3(256), 0, stream>>>(src, tgt, sq, nullptr, part1);
  mmd_bw<<<dim3(1), dim3(32), 0, stream>>>(part1, invbw);
  mmd_gemm<true><<<dim3(8, 8, 32), dim3(256), 0, stream>>>(src, tgt, sq, invbw, part3);
  mmd_final<<<dim3(1), dim3(32), 0, stream>>>(part3, out);
}
